// BidirectionalGRU_5488968205018
// MI455X (gfx1250) — matrix operand order repaired and checked by reading, not compiled, not run on hardware
//
#include <hip/hip_runtime.h>
#include <hip/hip_bf16.h>

// ---------------------------------------------------------------------------
// CDNA5 (gfx1250, wave32) bidirectional GRU.
//
// Pipeline:
//   1) cvt f32->f16 for x and all weight matrices (f16 WMMA, f32 accumulate)
//   2) gi = x @ W_ih^T + b_ih   (big WMMA GEMM, both directions)
//   3) persistent scan kernel: h kept in LDS, W_hh fragments pinned in
//      VGPRs for the whole scan (loop-invariant), gh = h @ W_hh^T via WMMA,
//      f32 gate math, BT=16 batch rows per workgroup (batch rows are
//      independent across the recurrence -> no cross-WG sync),
//      next-step gi rows prefetched (global_prefetch_b8)
//   4) repeat for layer 1, then final projection GEMM into d_out
// ---------------------------------------------------------------------------

typedef __attribute__((ext_vector_type(16))) _Float16 v16h;
typedef __attribute__((ext_vector_type(8)))  _Float16 v8h;
typedef __attribute__((ext_vector_type(4)))  _Float16 v4h;
typedef __attribute__((ext_vector_type(8)))  float    v8f;

static __device__ __forceinline__ v8f wmma_f32_f16(v16h a, v16h b, v8f c) {
  // (neg_a, A, neg_b, B, c_mod, C, reuse_a, reuse_b)
  return __builtin_amdgcn_wmma_f32_16x16x32_f16(false, a, false, b, (short)0, c,
                                                false, false);
}

// A-matrix fragment, 16x32 f16 (ISA 7.12.2): lane m=lane&15 holds row m.
// kg = lane>>4 selects K in [kg*16, kg*16+16) (contiguous, matches B-frag).
static __device__ __forceinline__ v16h load_a_frag(const _Float16* A, int lda,
                                                   int m0, int k0, int lane) {
  const int m  = lane & 15;
  const int kg = lane >> 4;
  const _Float16* p = A + (size_t)(m0 + m) * lda + k0 + kg * 16;
  v8h lo = *(const v8h*)(p);
  v8h hi = *(const v8h*)(p + 8);
  v16h r;
#pragma unroll
  for (int i = 0; i < 8; ++i) { r[i] = lo[i]; r[i + 8] = hi[i]; }
  return r;
}

// B-matrix fragment, 32x16 f16 = W^T tile where W is (N x K) row-major.
// Lane n=lane&15 holds column n = W row n; kg = lane>>4 selects K in
// [kg*16, kg*16+16): one contiguous 32-byte chunk of W's row.
static __device__ __forceinline__ v16h load_b_frag(const _Float16* W, int ldw,
                                                   int n0, int k0, int lane) {
  const int n  = lane & 15;
  const int kg = lane >> 4;
  return *(const v16h*)(W + (size_t)(n0 + n) * ldw + k0 + kg * 16);
}

static __device__ __forceinline__ float sigmoid_f(float x) {
  return 1.f / (1.f + __expf(-x));
}
static __device__ __forceinline__ float tanh_f(float x) {
  // stable: +/-inf exp handled, stays on v_exp_f32 fast path
  return 1.f - 2.f / (1.f + __expf(2.f * x));
}

// ---------------------------------------------------------------------------
// f32 -> f16 conversion, 4 elements/thread (all sizes are multiples of 4)
// ---------------------------------------------------------------------------
__global__ void cvt_f32_f16_kernel(const float* __restrict__ in,
                                   _Float16* __restrict__ out, long n4) {
  long i = (long)blockIdx.x * blockDim.x + threadIdx.x;
  const long stride = (long)gridDim.x * blockDim.x;
  const float4* in4 = (const float4*)in;
  v4h* out4 = (v4h*)out;
  for (; i < n4; i += stride) {
    float4 v = in4[i];
    v4h h;
    h[0] = (_Float16)v.x; h[1] = (_Float16)v.y;
    h[2] = (_Float16)v.z; h[3] = (_Float16)v.w;
    out4[i] = h;
  }
}

// ---------------------------------------------------------------------------
// C(M,N) = A(M,K) @ W(N,K)^T + bias(N), f16 in, f32 out.
// 256 threads = 8 waves; wave handles one 16-row M tile x three 16-col N
// tiles (48 cols). grid = (M/128, N/48). M%128==0, K%32==0, N%48==0.
// Branch-free accumulate loop (no N guards).
// ---------------------------------------------------------------------------
__global__ __launch_bounds__(256) void gemm_wt_bias_kernel(
    const _Float16* __restrict__ A, const _Float16* __restrict__ W,
    const float* __restrict__ bias, float* __restrict__ C,
    int M, int N, int K) {
  const int lane = threadIdx.x & 31;
  const int wave = threadIdx.x >> 5;
  const int m0 = (blockIdx.x * 8 + wave) * 16;
  const int nb = blockIdx.y * 48;
  if (m0 >= M) return;

  v8f acc[3] = {};
#pragma unroll 2
  for (int k0 = 0; k0 < K; k0 += 32) {
    v16h a = load_a_frag(A, K, m0, k0, lane);
#pragma unroll
    for (int u = 0; u < 3; ++u) {
      v16h b = load_b_frag(W, K, nb + u * 16, k0, lane);
      acc[u] = wmma_f32_f16(a, b, acc[u]);
    }
  }

  const int hi = lane >> 4, lo = lane & 15;
#pragma unroll
  for (int u = 0; u < 3; ++u) {
    const int n0 = nb + u * 16;
    const float bv = bias[n0 + lo];
#pragma unroll
    for (int j = 0; j < 8; ++j)
      C[(size_t)(m0 + hi * 8 + j) * N + n0 + lo] = acc[u][j] + bv;
  }
}

// ---------------------------------------------------------------------------
// Persistent GRU scan. One WG owns BT=16 batch rows for one direction.
// grid = (B/BT, 2): blockIdx.y = direction (0 fwd, 1 bwd).
// LDS: h (f32 + padded f16 copies) + padded gh staging tile  -> ~74 KB.
// Each wave pins its 3 N-tiles x 8 K-slices of W_hh (24 x v16h = 192 VGPRs)
// in registers for the entire scan: steady state has NO global loads in the
// GEMM phase. Per step: LDS A-frags + 24 WMMAs, barrier, f32 gates, barrier.
// Output written into (T, B, 2H) at column offset dir*H (layer concat).
// ---------------------------------------------------------------------------
constexpr int SCAN_BT      = 16;
constexpr int SCAN_THREADS = 512;   // 16 waves (4 per SIMD32)

__global__ __launch_bounds__(SCAN_THREADS) void gru_scan_kernel(
    const float* __restrict__ gi_f, const float* __restrict__ gi_b,
    const _Float16* __restrict__ whh_f, const _Float16* __restrict__ whh_b,
    const float* __restrict__ bhh_f, const float* __restrict__ bhh_b,
    float* __restrict__ out, int T, int Bsz) {
  constexpr int H   = 256;
  constexpr int G3  = 768;
  constexpr int HP  = H + 8;    // f16 row stride: 528 B => 4-bank rotation/row
  constexpr int G3P = G3 + 4;   // f32 row stride: de-conflicts rows r and r+8

  __shared__ __align__(32) _Float16 sh16[SCAN_BT * HP];   // h as f16 (padded)
  __shared__ float                  shf [SCAN_BT * H];    // h as f32
  __shared__ float                  sgh [SCAN_BT * G3P];  // gh staging (padded)

  const int dir = blockIdx.y;
  const int b0  = blockIdx.x * SCAN_BT;
  const float*    gi  = dir ? gi_b  : gi_f;
  const _Float16* whh = dir ? whh_b : whh_f;
  const float*    bhh = dir ? bhh_b : bhh_f;
  const int col_off   = dir ? H : 0;

  const int tid  = threadIdx.x;
  const int lane = tid & 31;
  const int wave = tid >> 5;       // 0..15; wave owns N tiles [3w, 3w+3)
  const int jb   = wave * 3;

  // ---- pin this wave's W_hh slice in VGPRs (loop-invariant) --------------
  v16h breg[24];                   // [kk*3 + u], kk = K/32 slice, u = N tile
#pragma unroll
  for (int kk = 0; kk < 8; ++kk)
#pragma unroll
    for (int u = 0; u < 3; ++u)
      breg[kk * 3 + u] = load_b_frag(whh, H, (jb + u) * 16, kk * 32, lane);

  // h0 = 0 (zero the padding too so WMMA A-frags never read garbage)
  for (int i = tid; i < SCAN_BT * HP; i += SCAN_THREADS) sh16[i] = (_Float16)0.f;
  for (int i = tid; i < SCAN_BT * H;  i += SCAN_THREADS) shf[i] = 0.f;
  __syncthreads();

  // gate-phase mapping: thread handles column n_ for 8 rows
  const int n_ = tid & (H - 1);
  const int rh = tid >> 8;                  // 0 or 1
  const float br = bhh[n_], bz = bhh[H + n_], bn = bhh[2 * H + n_];
  const int hi = lane >> 4, lo = lane & 15;

  for (int s = 0; s < T; ++s) {
    const int t = dir ? (T - 1 - s) : s;

    // ---- gh = h @ W_hh^T : (16 x 256) x (256 x 768), weights in VGPRs ----
    {
      v8f acc[3] = {};
#pragma unroll
      for (int kk = 0; kk < 8; ++kk) {
        v16h a = load_a_frag(sh16, HP, 0, kk * 32, lane);  // LDS, conflict-free
#pragma unroll
        for (int u = 0; u < 3; ++u)
          acc[u] = wmma_f32_f16(a, breg[kk * 3 + u], acc[u]);
      }
#pragma unroll
      for (int u = 0; u < 3; ++u)
#pragma unroll
        for (int j = 0; j < 8; ++j)
          sgh[(hi * 8 + j) * G3P + (jb + u) * 16 + lo] = acc[u][j];
    }
    __syncthreads();

    // ---- gates + state update (f32) ------------------------------------
    const float* git = gi + ((size_t)t * Bsz + b0) * G3;
#pragma unroll
    for (int i = 0; i < SCAN_BT / 2; ++i) {
      const int r = rh * (SCAN_BT / 2) + i;
      const float* girow = git + (size_t)r * G3;
      const float ir  = girow[n_];
      const float iz  = girow[H + n_];
      const float inn = girow[2 * H + n_];
      const float hr = sgh[r * G3P + n_] + br;
      const float hz = sgh[r * G3P + H + n_] + bz;
      const float hn = sgh[r * G3P + 2 * H + n_] + bn;
      const float rg = sigmoid_f(ir + hr);
      const float zg = sigmoid_f(iz + hz);
      const float ng = tanh_f(inn + rg * hn);
      const float hprev = shf[r * H + n_];
      const float hnew  = (1.f - zg) * ng + zg * hprev;
      shf[r * H + n_]   = hnew;
      sh16[r * HP + n_] = (_Float16)hnew;
      out[((size_t)t * Bsz + (b0 + r)) * (2 * H) + col_off + n_] = hnew;
    }

    // ---- prefetch next step's gi rows (hides the only global reads) -----
    if (s + 1 < T) {
      const int tn = dir ? (T - 2 - s) : (s + 1);
      const float* gin = gi + ((size_t)tn * Bsz + b0) * G3;
#pragma unroll
      for (int i = 0; i < SCAN_BT / 2; ++i) {
        const int r = rh * (SCAN_BT / 2) + i;
        const float* grow = gin + (size_t)r * G3;
        __builtin_prefetch(grow + n_, 0, 3);
        __builtin_prefetch(grow + H + n_, 0, 3);
        __builtin_prefetch(grow + 2 * H + n_, 0, 3);
      }
    }
    __syncthreads();
  }
}

// ---------------------------------------------------------------------------
// Host launch
// ---------------------------------------------------------------------------
extern "C" void kernel_launch(void* const* d_in, const int* in_sizes, int n_in,
                              void* d_out, int out_size, void* d_ws,
                              size_t ws_size, hipStream_t stream) {
  (void)in_sizes; (void)n_in; (void)out_size; (void)ws_size;
  constexpr int T = 256, B = 128, NIN = 256, H = 256, G3 = 768, NOUT = 96;
  constexpr long TB = (long)T * B;  // 32768

  const float* x        = (const float*)d_in[0];
  const float* w_ih_l0f = (const float*)d_in[1];
  const float* w_hh_l0f = (const float*)d_in[2];
  const float* b_ih_l0f = (const float*)d_in[3];
  const float* b_hh_l0f = (const float*)d_in[4];
  const float* w_ih_l0b = (const float*)d_in[5];
  const float* w_hh_l0b = (const float*)d_in[6];
  const float* b_ih_l0b = (const float*)d_in[7];
  const float* b_hh_l0b = (const float*)d_in[8];
  const float* w_ih_l1f = (const float*)d_in[9];
  const float* w_hh_l1f = (const float*)d_in[10];
  const float* b_ih_l1f = (const float*)d_in[11];
  const float* b_hh_l1f = (const float*)d_in[12];
  const float* w_ih_l1b = (const float*)d_in[13];
  const float* w_hh_l1b = (const float*)d_in[14];
  const float* b_ih_l1b = (const float*)d_in[15];
  const float* b_hh_l1b = (const float*)d_in[16];
  const float* w_emb    = (const float*)d_in[17];
  const float* b_emb    = (const float*)d_in[18];

  // ---- carve workspace (256 B aligned regions) ----------------------------
  char* wp = (char*)d_ws;
  auto carve = [&wp](size_t bytes) -> char* {
    char* r = wp;
    wp += (bytes + 255) & ~(size_t)255;
    return r;
  };
  _Float16* x16    = (_Float16*)carve(TB * NIN * sizeof(_Float16));
  _Float16* h1_16  = (_Float16*)carve(TB * 2 * H * sizeof(_Float16));
  _Float16* h2_16  = (_Float16*)carve(TB * 2 * H * sizeof(_Float16));
  float*    h1     = (float*)carve(TB * 2 * H * sizeof(float));
  float*    h2     = (float*)carve(TB * 2 * H * sizeof(float));
  float*    giA    = (float*)carve(TB * G3 * sizeof(float));  // reused l0/l1
  float*    giB    = (float*)carve(TB * G3 * sizeof(float));
  _Float16* wi0f16 = (_Float16*)carve((size_t)G3 * NIN * 2);
  _Float16* wi0b16 = (_Float16*)carve((size_t)G3 * NIN * 2);
  _Float16* wi1f16 = (_Float16*)carve((size_t)G3 * 2 * H * 2);
  _Float16* wi1b16 = (_Float16*)carve((size_t)G3 * 2 * H * 2);
  _Float16* wh0f16 = (_Float16*)carve((size_t)G3 * H * 2);
  _Float16* wh0b16 = (_Float16*)carve((size_t)G3 * H * 2);
  _Float16* wh1f16 = (_Float16*)carve((size_t)G3 * H * 2);
  _Float16* wh1b16 = (_Float16*)carve((size_t)G3 * H * 2);
  _Float16* wemb16 = (_Float16*)carve((size_t)NOUT * 2 * H * 2);

  auto cvt = [&](const float* src, _Float16* dst, long n) {
    const long n4 = n / 4;
    long blocks = (n4 + 255) / 256;
    if (blocks > 2048) blocks = 2048;
    cvt_f32_f16_kernel<<<dim3((unsigned)blocks), dim3(256), 0, stream>>>(src, dst, n4);
  };

  // ---- precision conversion ----------------------------------------------
  cvt(x, x16, TB * NIN);
  cvt(w_ih_l0f, wi0f16, (long)G3 * NIN);
  cvt(w_ih_l0b, wi0b16, (long)G3 * NIN);
  cvt(w_ih_l1f, wi1f16, (long)G3 * 2 * H);
  cvt(w_ih_l1b, wi1b16, (long)G3 * 2 * H);
  cvt(w_hh_l0f, wh0f16, (long)G3 * H);
  cvt(w_hh_l0b, wh0b16, (long)G3 * H);
  cvt(w_hh_l1f, wh1f16, (long)G3 * H);
  cvt(w_hh_l1b, wh1b16, (long)G3 * H);
  cvt(w_emb,    wemb16, (long)NOUT * 2 * H);

  const dim3 gblk(256);
  const dim3 gi_grid(TB / 128, G3 / 48);       // (256, 16)

  // ---- layer 0 ------------------------------------------------------------
  gemm_wt_bias_kernel<<<gi_grid, gblk, 0, stream>>>(x16, wi0f16, b_ih_l0f, giA,
                                                    (int)TB, G3, NIN);
  gemm_wt_bias_kernel<<<gi_grid, gblk, 0, stream>>>(x16, wi0b16, b_ih_l0b, giB,
                                                    (int)TB, G3, NIN);
  gru_scan_kernel<<<dim3(B / SCAN_BT, 2), dim3(SCAN_THREADS), 0, stream>>>(
      giA, giB, wh0f16, wh0b16, b_hh_l0f, b_hh_l0b, h1, T, B);

  // ---- layer 1 ------------------------------------------------------------
  cvt(h1, h1_16, TB * 2 * H);
  gemm_wt_bias_kernel<<<gi_grid, gblk, 0, stream>>>(h1_16, wi1f16, b_ih_l1f, giA,
                                                    (int)TB, G3, 2 * H);
  gemm_wt_bias_kernel<<<gi_grid, gblk, 0, stream>>>(h1_16, wi1b16, b_ih_l1b, giB,
                                                    (int)TB, G3, 2 * H);
  gru_scan_kernel<<<dim3(B / SCAN_BT, 2), dim3(SCAN_THREADS), 0, stream>>>(
      giA, giB, wh1f16, wh1b16, b_hh_l1f, b_hh_l1b, h2, T, B);

  // ---- final projection ---------------------------------------------------
  cvt(h2, h2_16, TB * 2 * H);
  const dim3 out_grid(TB / 128, NOUT / 48);    // (256, 2)
  gemm_wt_bias_kernel<<<out_grid, gblk, 0, stream>>>(h2_16, wemb16, b_emb,
                                                     (float*)d_out,
                                                     (int)TB, NOUT, 2 * H);
}
